// GatedDeltaNet_81776177316019
// MI455X (gfx1250) — matrix operand order repaired and checked by reading, not compiled, not run on hardware
//
#include <hip/hip_runtime.h>
#include <cstdint>
#include <cstddef>

// ---------------------------------------------------------------------------
// Gated DeltaNet forward for MI455X (gfx1250, wave32, WMMA + async-to-LDS).
// B=2, T=4096, D=1024, H=16, DK=DV=64.
// ---------------------------------------------------------------------------

typedef __attribute__((ext_vector_type(16))) __bf16 bf16x16;
typedef __attribute__((ext_vector_type(8)))  __bf16 bf16x8;
typedef __attribute__((ext_vector_type(8)))  float  f32x8;

#define BATCH 2
#define SEQT  4096
#define DMODEL 1024
#define NHEAD 16
#define HDIM  64
#define ROWS  (BATCH * SEQT)          // 8192
#define NE_X  ((size_t)ROWS * DMODEL) // 8388608

// ---------------------------------------------------------------------------
// LDS helpers for async global->LDS copies (ASYNCcnt tracked).
// Low 32 bits of a generic shared pointer are the LDS byte offset (flat
// aperture truncation, ISA 10.2).
// ---------------------------------------------------------------------------
__device__ __forceinline__ uint32_t lds_addr_of(const void* p) {
    return (uint32_t)(uintptr_t)p;
}
__device__ __forceinline__ void async_copy_b128(uint32_t lds, const void* gaddr) {
    asm volatile("global_load_async_to_lds_b128 %0, %1, off"
                 :: "v"(lds), "v"((unsigned long long)(uintptr_t)gaddr)
                 : "memory");
}
__device__ __forceinline__ void wait_async0() {
    asm volatile("s_wait_asynccnt 0" ::: "memory");
}

// ---------------------------------------------------------------------------
// f32 -> bf16 convert (plain)
// ---------------------------------------------------------------------------
__global__ __launch_bounds__(256) void cvt_bf16_kernel(
    const float* __restrict__ in, __bf16* __restrict__ out, size_t n)
{
    size_t i = (size_t)blockIdx.x * blockDim.x + threadIdx.x;
    if (i < n) out[i] = (__bf16)in[i];
}

// ---------------------------------------------------------------------------
// f32 [K,N] -> bf16 [N,K] transpose-convert (32x32 LDS tile, coalesced).
// K,N multiples of 32.
// ---------------------------------------------------------------------------
__global__ __launch_bounds__(256) void cvt_transpose_bf16_kernel(
    const float* __restrict__ in, __bf16* __restrict__ out, int K, int N)
{
    __shared__ float tile[32][33];
    const int bn = blockIdx.x * 32;    // N tile
    const int bk = blockIdx.y * 32;    // K tile
    const int tx = threadIdx.x & 31;
    const int ty = threadIdx.x >> 5;   // 0..7
    #pragma unroll
    for (int i = ty; i < 32; i += 8)
        tile[i][tx] = in[(size_t)(bk + i) * N + bn + tx];
    __syncthreads();
    #pragma unroll
    for (int i = ty; i < 32; i += 8)
        out[(size_t)(bn + i) * K + bk + tx] = (__bf16)tile[tx][i];
}

// ---------------------------------------------------------------------------
// Tiled bf16 WMMA GEMM: C[M,N] = act(A[M,K] @ B[K,N]) with B supplied
// TRANSPOSED (Bt[N,K], row-major). f32 out.
// Workgroup tile 64(M) x 128(N), 8 waves -> each wave 32x32 (2x2 WMMA),
// K-step 32, double-buffered LDS filled by global_load_async_to_lds_b128.
// ACT: 0=none, 1=silu, 2=sigmoid.  M%64==0, N%128==0, K%32==0.
// ---------------------------------------------------------------------------
template <int ACT>
__global__ __launch_bounds__(256) void gemm_bf16_wmma(
    const __bf16* __restrict__ A, const __bf16* __restrict__ Bt,
    float* __restrict__ C, int M, int N, int K)
{
    __shared__ __align__(16) __bf16 As[2][64 * 32];    // [m][k]
    __shared__ __align__(16) __bf16 Bs[2][128 * 32];   // [n][k]

    const int tid  = threadIdx.x;
    const int wave = tid >> 5;
    const int lane = tid & 31;
    const int wm   = wave >> 2;        // 0..1
    const int wn   = wave & 3;         // 0..3
    const int hi   = lane >> 4;        // lane half
    const int l16  = lane & 15;
    const int tile_m = blockIdx.y * 64;
    const int tile_n = blockIdx.x * 128;

    // Per-thread staging coordinates (all 16B-contiguous chunks).
    const int ar = tid >> 2;           // A row 0..63
    const int ac = (tid & 3) * 8;      // A col {0,8,16,24}

    auto issue = [&](int buf, int k0) {
        // A tile 64x32: one b128 per thread
        async_copy_b128(lds_addr_of(&As[buf][ar * 32 + ac]),
                        &A[(size_t)(tile_m + ar) * K + k0 + ac]);
        // B tile 128x32 (from Bt rows): two b128 per thread
        #pragma unroll
        for (int p = 0; p < 2; ++p) {
            int qq = tid * 2 + p;      // chunk 0..511
            int n  = qq >> 2;
            int kc = (qq & 3) * 8;
            async_copy_b128(lds_addr_of(&Bs[buf][n * 32 + kc]),
                            &Bt[(size_t)(tile_n + n) * K + k0 + kc]);
        }
    };

    f32x8 acc[2][2] = {};
    issue(0, 0);

    const int nk = K / 32;
    for (int it = 0; it < nk; ++it) {
        const int cur = it & 1;
        wait_async0();                 // my copies for buffer `cur` complete
        __syncthreads();               // everyone's copies visible; prev reads done
        if (it + 1 < nk) issue(cur ^ 1, (it + 1) * 32);  // overlap with compute

        // ---- fragments per ISA 16-bit layouts ----
        // A 16x32: lane<16 row M=l16 holds K {0..7,16..23}; lane>=16 shifted +8.
        bf16x16 afr[2];
        #pragma unroll
        for (int mi = 0; mi < 2; ++mi) {
            const __bf16* ap = &As[cur][(wm * 32 + mi * 16 + l16) * 32 + hi * 8];
            bf16x8 lo = *reinterpret_cast<const bf16x8*>(ap);
            bf16x8 hh = *reinterpret_cast<const bf16x8*>(ap + 16);
            bf16x16 a;
            #pragma unroll
            for (int e = 0; e < 8; ++e) { a[e] = lo[e]; a[e + 8] = hh[e]; }
            afr[mi] = a;
        }
        // B 32x16: lane column N=l16, K {0..7,16..23} for lane<16; lane>=16 shifted +8.
        bf16x16 bfr[2];
        #pragma unroll
        for (int ni = 0; ni < 2; ++ni) {
            const __bf16* bp = &Bs[cur][(wn * 32 + ni * 16 + l16) * 32 + hi * 8];
            bf16x8 lo = *reinterpret_cast<const bf16x8*>(bp);
            bf16x8 hh = *reinterpret_cast<const bf16x8*>(bp + 16);
            bf16x16 b;
            #pragma unroll
            for (int e = 0; e < 8; ++e) { b[e] = lo[e]; b[e + 8] = hh[e]; }
            bfr[ni] = b;
        }

        #pragma unroll
        for (int mi = 0; mi < 2; ++mi)
            #pragma unroll
            for (int ni = 0; ni < 2; ++ni)
                acc[mi][ni] = __builtin_amdgcn_wmma_f32_16x16x32_bf16(
                    false, afr[mi], false, bfr[ni], (short)0, acc[mi][ni],
                    false, false);
        // no trailing barrier: next iteration's wait+barrier covers the hazard
    }

    // ---- epilogue: C layout VGPR r -> M = r + 8*hi, N = l16 ----
    #pragma unroll
    for (int mi = 0; mi < 2; ++mi) {
        #pragma unroll
        for (int ni = 0; ni < 2; ++ni) {
            #pragma unroll
            for (int r = 0; r < 8; ++r) {
                int m = tile_m + wm * 32 + mi * 16 + r + hi * 8;
                int n = tile_n + wn * 32 + ni * 16 + l16;
                float v = acc[mi][ni][r];
                if constexpr (ACT == 1) {              // silu
                    v = v * (1.f / (1.f + __expf(-v)));
                } else if constexpr (ACT == 2) {       // sigmoid
                    v = 1.f / (1.f + __expf(-v));
                }
                C[(size_t)m * N + n] = v;
            }
        }
    }
}

// ---------------------------------------------------------------------------
// Causal depthwise conv1d (K=4) + SiLU.  in/out: [B,T,C], w: [C,4], b: [C].
// ---------------------------------------------------------------------------
__global__ __launch_bounds__(256) void dwconv_silu_kernel(
    const float* __restrict__ in, const float* __restrict__ w,
    const float* __restrict__ bias, float* __restrict__ out, size_t n)
{
    size_t idx = (size_t)blockIdx.x * blockDim.x + threadIdx.x;
    if (idx >= n) return;
    int c = (int)(idx % DMODEL);
    int t = (int)((idx / DMODEL) % SEQT);
    float acc = bias[c];
    #pragma unroll
    for (int kk = 0; kk < 4; ++kk) {
        int tt = t - 3 + kk;
        float xv = (tt >= 0) ? in[(long long)idx + (long long)(kk - 3) * DMODEL] : 0.f;
        acc = fmaf(xv, w[c * 4 + kk], acc);
    }
    out[idx] = acc * (1.f / (1.f + __expf(-acc)));
}

// ---------------------------------------------------------------------------
// In-place L2 normalization of contiguous groups of 64 (per b,t,h head slice).
// ---------------------------------------------------------------------------
__global__ __launch_bounds__(256) void l2norm64_kernel(
    float* __restrict__ t, size_t n_groups)
{
    size_t g = (size_t)blockIdx.x * blockDim.x + threadIdx.x;
    if (g >= n_groups) return;
    float* p = t + g * HDIM;
    float s = 0.f;
    #pragma unroll 8
    for (int i = 0; i < HDIM; ++i) s = fmaf(p[i], p[i], s);
    float inv = 1.f / fmaxf(sqrtf(s), 1e-6f);
    #pragma unroll 8
    for (int i = 0; i < HDIM; ++i) p[i] *= inv;
}

// ---------------------------------------------------------------------------
// alpha/beta gates: per (row, h) dot over D. alpha=clip(sig,0.1,1), beta=min(sig,1).
// ---------------------------------------------------------------------------
__global__ __launch_bounds__(256) void gates_kernel(
    const float* __restrict__ x, const float* __restrict__ Wa,
    const float* __restrict__ ba, const float* __restrict__ Wb,
    const float* __restrict__ bbv, float* __restrict__ alpha,
    float* __restrict__ beta, int rows)
{
    int idx = blockIdx.x * blockDim.x + threadIdx.x;
    if (idx >= rows * NHEAD) return;
    int h = idx & (NHEAD - 1);
    int r = idx >> 4;
    const float* xr = x + (size_t)r * DMODEL;
    float sa = ba[h], sb = bbv[h];
    for (int i = 0; i < DMODEL; ++i) {
        float xv = xr[i];
        sa = fmaf(xv, Wa[i * NHEAD + h], sa);
        sb = fmaf(xv, Wb[i * NHEAD + h], sb);
    }
    float av = 1.f / (1.f + __expf(-sa));
    av = fminf(fmaxf(av, 0.1f), 1.0f);
    float bv = 1.f / (1.f + __expf(-sb));
    bv = fminf(bv, 1.0f);
    alpha[idx] = av;
    beta[idx]  = bv;
}

// ---------------------------------------------------------------------------
// Gated delta-rule recurrence. One block per (b,h); 64 threads; thread j owns
// column j of the 64x64 state S in VGPRs. k/q broadcast through LDS per step.
//   v_hat = S^T k ; err = v - v_hat ; S = a*S + b*(k outer err) ; o = S^T q
// q/k/v/O layouts: [B,T,H*64]; alpha/beta: [B*T,H].
// ---------------------------------------------------------------------------
__global__ __launch_bounds__(64) void delta_recurrence_kernel(
    const float* __restrict__ q, const float* __restrict__ k,
    const float* __restrict__ v, const float* __restrict__ alpha,
    const float* __restrict__ beta, float* __restrict__ O)
{
    const int bh = blockIdx.x;           // 0..31
    const int b  = bh >> 4;
    const int h  = bh & (NHEAD - 1);
    const int j  = threadIdx.x;          // 0..63

    __shared__ float ksh[HDIM];
    __shared__ float qsh[HDIM];

    float S[HDIM];
    #pragma unroll
    for (int i = 0; i < HDIM; ++i) S[i] = 0.f;

    const size_t base = (size_t)b * SEQT * DMODEL + (size_t)h * HDIM;

    for (int t = 0; t < SEQT; ++t) {
        const size_t off = base + (size_t)t * DMODEL;
        ksh[j] = k[off + j];
        qsh[j] = q[off + j];
        const float vj = v[off + j];
        const size_t gidx = ((size_t)b * SEQT + t) * NHEAD + h;
        const float a  = alpha[gidx];
        const float bt = beta[gidx];
        __syncthreads();

        float vh = 0.f;
        #pragma unroll
        for (int i = 0; i < HDIM; ++i) vh = fmaf(S[i], ksh[i], vh);
        const float err = (vj - vh) * bt;

        float oj = 0.f;
        #pragma unroll
        for (int i = 0; i < HDIM; ++i) {
            S[i] = fmaf(S[i], a, ksh[i] * err);
            oj = fmaf(S[i], qsh[i], oj);
        }
        O[off + j] = oj;
        __syncthreads();
    }
}

// ---------------------------------------------------------------------------
// RMSNorm (per 64-group) * rms_w * sigmoid-gate; emit bf16 for the final GEMM.
// ---------------------------------------------------------------------------
__global__ __launch_bounds__(256) void rmsnorm_gate_kernel(
    const float* __restrict__ O, const float* __restrict__ g,
    const float* __restrict__ rms_w, __bf16* __restrict__ out, size_t n_groups)
{
    size_t gi = (size_t)blockIdx.x * blockDim.x + threadIdx.x;
    if (gi >= n_groups) return;
    int h = (int)(gi & (NHEAD - 1));     // group index = (b*T+t)*H + h
    const float* op = O + gi * HDIM;
    const float* gp = g + gi * HDIM;
    float ss = 0.f;
    #pragma unroll 8
    for (int i = 0; i < HDIM; ++i) ss = fmaf(op[i], op[i], ss);
    float r = rsqrtf(ss * (1.0f / HDIM) + 1e-6f);
    #pragma unroll 8
    for (int i = 0; i < HDIM; ++i)
        out[gi * HDIM + i] = (__bf16)(op[i] * r * rms_w[h * HDIM + i] * gp[i]);
}

// ---------------------------------------------------------------------------
// Launch chain
// ---------------------------------------------------------------------------
extern "C" void kernel_launch(void* const* d_in, const int* in_sizes, int n_in,
                              void* d_out, int out_size, void* d_ws, size_t ws_size,
                              hipStream_t stream)
{
    (void)in_sizes; (void)n_in; (void)out_size; (void)ws_size;

    const float* x    = (const float*)d_in[0];
    const float* Wq   = (const float*)d_in[1];
    const float* Wk   = (const float*)d_in[2];
    const float* Wv   = (const float*)d_in[3];
    const float* Wo   = (const float*)d_in[4];
    const float* Wa   = (const float*)d_in[5];
    const float* ba   = (const float*)d_in[6];
    const float* Wb   = (const float*)d_in[7];
    const float* bb   = (const float*)d_in[8];
    const float* Wgd  = (const float*)d_in[9];
    const float* Wgu  = (const float*)d_in[10];
    const float* rmsw = (const float*)d_in[11];
    const float* qcw  = (const float*)d_in[12];
    const float* qcb  = (const float*)d_in[13];
    const float* kcw  = (const float*)d_in[14];
    const float* kcb  = (const float*)d_in[15];
    const float* vcw  = (const float*)d_in[16];
    const float* vcb  = (const float*)d_in[17];

    const size_t NE_W   = (size_t)DMODEL * DMODEL;      // 1M
    const size_t NE_GD  = (size_t)DMODEL * 512;         // Wgd
    const size_t NE_GU  = (size_t)512 * DMODEL;         // Wgu
    const size_t NGROUP = (size_t)ROWS * NHEAD;         // 131072

    char* ws = (char*)d_ws;
    size_t off = 0;
    auto alloc = [&](size_t bytes) -> char* {
        char* p = ws + off;
        off = (off + bytes + 255) & ~(size_t)255;
        return p;
    };

    __bf16* x_bf   = (__bf16*)alloc(NE_X * 2);
    __bf16* wq_t   = (__bf16*)alloc(NE_W * 2);     // [N,K] transposed bf16
    __bf16* wk_t   = (__bf16*)alloc(NE_W * 2);
    __bf16* wv_t   = (__bf16*)alloc(NE_W * 2);
    __bf16* wo_t   = (__bf16*)alloc(NE_W * 2);
    __bf16* wgd_t  = (__bf16*)alloc(NE_GD * 2);
    __bf16* wgu_t  = (__bf16*)alloc(NE_GU * 2);
    float*  qp     = (float*)alloc(NE_X * 4);
    float*  kp     = (float*)alloc(NE_X * 4);
    float*  vp     = (float*)alloc(NE_X * 4);
    float*  qc     = (float*)alloc(NE_X * 4);
    float*  kc     = (float*)alloc(NE_X * 4);
    float*  vc     = (float*)alloc(NE_X * 4);
    float*  alp    = (float*)alloc(NGROUP * 4);
    float*  bet    = (float*)alloc(NGROUP * 4);
    float*  xg     = (float*)alloc((size_t)ROWS * 512 * 4);
    __bf16* xg_bf  = (__bf16*)alloc((size_t)ROWS * 512 * 2);
    float*  gbuf   = (float*)alloc(NE_X * 4);
    float*  Obuf   = (float*)alloc(NE_X * 4);
    __bf16* go_bf  = (__bf16*)alloc(NE_X * 2);

    const int TPB = 256;
    auto blocks = [](size_t n, int t) { return (unsigned)((n + t - 1) / t); };

    // 1) conversions: activations plain, weights transpose-convert to [N,K]
    cvt_bf16_kernel<<<blocks(NE_X, TPB), TPB, 0, stream>>>(x, x_bf, NE_X);
    dim3 t1024(DMODEL / 32, DMODEL / 32);
    cvt_transpose_bf16_kernel<<<t1024, 256, 0, stream>>>(Wq, wq_t, DMODEL, DMODEL);
    cvt_transpose_bf16_kernel<<<t1024, 256, 0, stream>>>(Wk, wk_t, DMODEL, DMODEL);
    cvt_transpose_bf16_kernel<<<t1024, 256, 0, stream>>>(Wv, wv_t, DMODEL, DMODEL);
    cvt_transpose_bf16_kernel<<<t1024, 256, 0, stream>>>(Wo, wo_t, DMODEL, DMODEL);
    cvt_transpose_bf16_kernel<<<dim3(512 / 32, DMODEL / 32), 256, 0, stream>>>(Wgd, wgd_t, DMODEL, 512);
    cvt_transpose_bf16_kernel<<<dim3(DMODEL / 32, 512 / 32), 256, 0, stream>>>(Wgu, wgu_t, 512, DMODEL);

    // 2) q/k/v projections (WMMA, async-LDS double-buffered)
    dim3 g1024(DMODEL / 128, ROWS / 64);
    gemm_bf16_wmma<0><<<g1024, 256, 0, stream>>>(x_bf, wq_t, qp, ROWS, DMODEL, DMODEL);
    gemm_bf16_wmma<0><<<g1024, 256, 0, stream>>>(x_bf, wk_t, kp, ROWS, DMODEL, DMODEL);
    gemm_bf16_wmma<0><<<g1024, 256, 0, stream>>>(x_bf, wv_t, vp, ROWS, DMODEL, DMODEL);

    // 3) causal depthwise conv + SiLU
    dwconv_silu_kernel<<<blocks(NE_X, TPB), TPB, 0, stream>>>(qp, qcw, qcb, qc, NE_X);
    dwconv_silu_kernel<<<blocks(NE_X, TPB), TPB, 0, stream>>>(kp, kcw, kcb, kc, NE_X);
    dwconv_silu_kernel<<<blocks(NE_X, TPB), TPB, 0, stream>>>(vp, vcw, vcb, vc, NE_X);

    // 4) L2 norm q, k per head slice
    l2norm64_kernel<<<blocks(NGROUP, TPB), TPB, 0, stream>>>(qc, NGROUP);
    l2norm64_kernel<<<blocks(NGROUP, TPB), TPB, 0, stream>>>(kc, NGROUP);

    // 5) alpha/beta gates
    gates_kernel<<<blocks(NGROUP, TPB), TPB, 0, stream>>>(x, Wa, ba, Wb, bb, alp, bet, ROWS);

    // 6) output-gate path: silu(x@Wgd) -> bf16 -> sigmoid(@Wgu)
    dim3 g512(512 / 128, ROWS / 64);
    gemm_bf16_wmma<1><<<g512, 256, 0, stream>>>(x_bf, wgd_t, xg, ROWS, 512, DMODEL);
    cvt_bf16_kernel<<<blocks((size_t)ROWS * 512, TPB), TPB, 0, stream>>>(xg, xg_bf, (size_t)ROWS * 512);
    gemm_bf16_wmma<2><<<g1024, 256, 0, stream>>>(xg_bf, wgu_t, gbuf, ROWS, DMODEL, 512);

    // 7) sequential gated delta-rule recurrence
    delta_recurrence_kernel<<<BATCH * NHEAD, HDIM, 0, stream>>>(qc, kc, vc, alp, bet, Obuf);

    // 8) RMSNorm * rms_w * gate -> bf16
    rmsnorm_gate_kernel<<<blocks(NGROUP, TPB), TPB, 0, stream>>>(Obuf, gbuf, rmsw, go_bf, NGROUP);

    // 9) final projection (WMMA) into d_out
    gemm_bf16_wmma<0><<<g1024, 256, 0, stream>>>(go_bf, wo_t, (float*)d_out, ROWS, DMODEL, DMODEL);
}
